// RGAT_10763188044024
// MI455X (gfx1250) — compile-verified, matrix operand order checked by reading, not run on hardware
//
#include <hip/hip_runtime.h>

// ---- problem constants (match reference) ----
#define N_SRC0_ 120000
#define N_DST0_ 40000
#define N_DST1_ 10000
#define T_      5
#define H_      4
#define HID_    128
#define D_      32
#define IN_     256
#define OUT_    153
#define E0_     800000
#define E1_     300000
#define NEG_SLOPE_ 0.2f
#define EPS_       1e-5f

typedef __attribute__((ext_vector_type(16))) __bf16 v16bf;
typedef __attribute__((ext_vector_type(8)))  float  v8f;

#define MAX_KT_ 8    // max K/32 (K=256)
#define APAD_   24   // padded per-lane fragment stride (elements, 48B -> bank spread)

// ---------- helpers ----------
__device__ __forceinline__ unsigned ordKey(float f) {
  unsigned u = __float_as_uint(f);
  return (u & 0x80000000u) ? ~u : (u | 0x80000000u);
}
__device__ __forceinline__ float unordKey(unsigned k) {
  return (k & 0x80000000u) ? __uint_as_float(k ^ 0x80000000u)
                           : __uint_as_float(~k);
}

__global__ void fill_u32(unsigned* __restrict__ p, long n, unsigned v) {
  long i = (long)blockIdx.x * blockDim.x + threadIdx.x;
  long step = (long)gridDim.x * blockDim.x;
  for (; i < n; i += step) p[i] = v;
}

// Pre-split weight matrix (KxN, row-major, f32) into bf16 hi/lo B-fragments.
// Fragment layout per (nt, kt): 32 lanes x 16 elements, contiguous per lane.
// lane -> column (lane&15), lane bit4 -> K half; element t -> K = kb + t (+8 if t>=8).
__global__ void make_frags(const float* __restrict__ W, int K, int N,
                           int ktiles, int ntiles,
                           unsigned short* __restrict__ hi,
                           unsigned short* __restrict__ lo) {
  int total = ntiles * ktiles * 512;
  int idx = blockIdx.x * blockDim.x + threadIdx.x;
  if (idx >= total) return;
  int t    = idx & 15;
  int lane = (idx >> 4) & 31;
  int kt   = (idx >> 9) % ktiles;
  int nt   = idx / (ktiles << 9);
  int n = nt * 16 + (lane & 15);
  int k = kt * 32 + ((lane & 16) ? 8 : 0) + t + ((t & 8) ? 8 : 0);
  float v = (n < N) ? W[(long)k * N + n] : 0.f;
  unsigned u  = __float_as_uint(v);
  unsigned uh = u & 0xFFFF0000u;
  float    r  = v - __uint_as_float(uh);
  hi[idx] = (unsigned short)(uh >> 16);
  lo[idx] = (unsigned short)(__float_as_uint(r) >> 16);
}

// bias_total[c] = bs[c] + sum_j b[j][c]   (folds per-type additive bias)
__global__ void bias_total_kernel(const float* __restrict__ bs,
                                  const float* __restrict__ b,
                                  float* __restrict__ outb) {
  int c = threadIdx.x;
  float v = bs[c];
  for (int j = 0; j < T_; ++j) v += b[j * HID_ + c];
  outb[c] = v;
}

// ---------- WMMA GEMM:  C(MxN) = A(MxK) @ W + bias  (bf16x3 split, fp32 acc) ----------
// Block = 256 threads (8 waves) owns one 16-row tile. The block cooperatively splits
// the A tile into bf16 hi/lo fragments in LDS ONCE; each wave then computes one
// 16-wide column tile reading A from LDS (ds_load_b128) and pre-split W fragments
// from global. Inner loop: 4 LDS b128 + 4 global b128 + 3 v_wmma.
__global__ __launch_bounds__(256)
void gemm_wmma_bf16x3(const float* __restrict__ A, int K,
                      const unsigned short* __restrict__ fragHi,
                      const unsigned short* __restrict__ fragLo,
                      int ntiles, int N,
                      const float* __restrict__ bias,
                      float* __restrict__ C) {
  __shared__ unsigned short sHi[MAX_KT_ * 32 * APAD_];
  __shared__ unsigned short sLo[MAX_KT_ * 32 * APAD_];
  const int tid    = threadIdx.x;
  const int lane   = tid & 31;
  const int wid    = tid >> 5;
  const int rt     = blockIdx.x;
  const int ktiles = K >> 5;

  // ---- cooperative stage: split A tile (16 rows x K) into LDS fragments ----
  const int total = ktiles << 9;            // ktiles * 512 elements
  for (int idx = tid; idx < total; idx += 256) {
    int t  = idx & 15;
    int ln = (idx >> 4) & 31;
    int kt = idx >> 9;
    int row = rt * 16 + (ln & 15);
    int k = kt * 32 + ((ln & 16) ? 8 : 0) + t + ((t & 8) ? 8 : 0);
    float v = A[(long)row * K + k];
    unsigned u  = __float_as_uint(v);
    unsigned uh = u & 0xFFFF0000u;
    float    r  = v - __uint_as_float(uh);
    int so = (kt * 32 + ln) * APAD_ + t;
    sHi[so] = (unsigned short)(uh >> 16);
    sLo[so] = (unsigned short)(__float_as_uint(r) >> 16);
  }
  __syncthreads();

  const int nt = blockIdx.y * 8 + wid;
  if (nt < ntiles) {                        // wave-uniform: EXEC all-ones for WMMA
    union Frag { v16bf v; uint4 q[2]; };
    v8f acc = {};
    const long fb = (long)nt * ktiles * 512 + lane * 16;
    for (int kt = 0; kt < ktiles; ++kt) {
      const int ao = (kt * 32 + lane) * APAD_;
      Frag ah, av, bh, bl;
      ah.q[0] = *(const uint4*)(sHi + ao);
      ah.q[1] = *(const uint4*)(sHi + ao + 8);
      av.q[0] = *(const uint4*)(sLo + ao);
      av.q[1] = *(const uint4*)(sLo + ao + 8);
      const unsigned short* bp = fragHi + fb + (long)kt * 512;
      const unsigned short* lp = fragLo + fb + (long)kt * 512;
      bh.q[0] = *(const uint4*)(bp);
      bh.q[1] = *(const uint4*)(bp + 8);
      bl.q[0] = *(const uint4*)(lp);
      bl.q[1] = *(const uint4*)(lp + 8);
      acc = __builtin_amdgcn_wmma_f32_16x16x32_bf16(false, ah.v, false, bh.v, (short)0, acc, false, false);
      acc = __builtin_amdgcn_wmma_f32_16x16x32_bf16(false, av.v, false, bh.v, (short)0, acc, false, false);
      acc = __builtin_amdgcn_wmma_f32_16x16x32_bf16(false, ah.v, false, bl.v, (short)0, acc, false, false);
    }
    const int cc = nt * 16 + (lane & 15);
    if (cc < N) {                           // divergence only after WMMA
      const float b = bias ? bias[cc] : 0.f;
      const int rbase = rt * 16 + ((lane & 16) ? 8 : 0);
#pragma unroll
      for (int m = 0; m < 8; ++m) C[(long)(rbase + m) * N + cc] = acc[m] + b;
    }
  }
}

// ---------- attention scalars:  el = <h_row, al[head]>, er = <h_row, ar[head]> ----------
__global__ void el_er_kernel(const float* __restrict__ h, int n_src, int n_dst,
                             const float* __restrict__ al, const float* __restrict__ ar,
                             float* __restrict__ el, float* __restrict__ er) {
  int idx = blockIdx.x * blockDim.x + threadIdx.x;
  if (idx >= n_src * H_) return;
  int i = idx >> 2, hd = idx & 3;
  const float* row = h + (long)i * HID_ + hd * D_;
  const float* a = al + hd * D_;
  float s = 0.f;
#pragma unroll
  for (int d = 0; d < D_; ++d) s += row[d] * a[d];
  el[idx] = s;
  if (i < n_dst) {
    const float* b = ar + hd * D_;
    float s2 = 0.f;
#pragma unroll
    for (int d = 0; d < D_; ++d) s2 += row[d] * b[d];
    er[idx] = s2;
  }
}

// ---------- edge pass 1: per-(dst,head) running max via ordered-uint atomicMax ----------
__global__ void edge_max_kernel(const int* __restrict__ src, const int* __restrict__ dst, int E,
                                const float* __restrict__ el, const float* __restrict__ er,
                                unsigned* __restrict__ mOrd) {
  int e = blockIdx.x * blockDim.x + threadIdx.x;
  if (e >= E) return;
  int s = src[e], d = dst[e];
#pragma unroll
  for (int hd = 0; hd < H_; ++hd) {
    float v = el[s * H_ + hd] + er[d * H_ + hd];
    v = (v >= 0.f) ? v : NEG_SLOPE_ * v;
    atomicMax(&mOrd[d * H_ + hd], ordKey(v));
  }
}

// ---------- edge pass 2: exp(e - max), store per-edge, accumulate denominator ----------
__global__ void edge_sum_kernel(const int* __restrict__ src, const int* __restrict__ dst, int E,
                                const float* __restrict__ el, const float* __restrict__ er,
                                const unsigned* __restrict__ mOrd,
                                float* __restrict__ exbuf, float* __restrict__ ssum) {
  int e = blockIdx.x * blockDim.x + threadIdx.x;
  if (e >= E) return;
  int s = src[e], d = dst[e];
#pragma unroll
  for (int hd = 0; hd < H_; ++hd) {
    float v = el[s * H_ + hd] + er[d * H_ + hd];
    v = (v >= 0.f) ? v : NEG_SLOPE_ * v;
    float m  = unordKey(mOrd[d * H_ + hd]);
    float ex = __expf(v - m);
    exbuf[(long)e * H_ + hd] = ex;
    atomicAdd(&ssum[d * H_ + hd], ex);
  }
}

// ---------- edge pass 3: one wave per edge, scatter a*h_src row (128 f32) ----------
__global__ void edge_msg_kernel(const int* __restrict__ src, const int* __restrict__ dst, int E,
                                const float* __restrict__ exbuf, const float* __restrict__ ssum,
                                const float* __restrict__ h, float* __restrict__ out) {
  long gid = (long)blockIdx.x * blockDim.x + threadIdx.x;
  int e = (int)(gid >> 5);
  if (e >= E) return;
  int lane = threadIdx.x & 31;
  int s = src[e], d = dst[e];
  int col = lane * 4;                 // 4 consecutive cols, all in same head
  int hd  = col >> 5;
  float a = exbuf[(long)e * H_ + hd] / (ssum[d * H_ + hd] + 1e-16f);
  float4 hv = *(const float4*)(h + (long)s * HID_ + col);
  float* o = out + (long)d * HID_ + col;
  atomicAdd(o + 0, a * hv.x);
  atomicAdd(o + 1, a * hv.y);
  atomicAdd(o + 2, a * hv.z);
  atomicAdd(o + 3, a * hv.w);
}

// ---------- per-column stats over rows (feature-wise mean/var) ----------
__global__ void colstats_kernel(const float* __restrict__ x, int n,
                                float* __restrict__ sums, float* __restrict__ sumsq) {
  int c = threadIdx.x;                // 128 threads = 128 columns
  float s = 0.f, q = 0.f;
  for (int r = blockIdx.x; r < n; r += gridDim.x) {
    float v = x[(long)r * HID_ + c];
    s += v; q += v * v;
  }
  atomicAdd(&sums[c], s);
  atomicAdd(&sumsq[c], q);
}

// actMode: 0 = ELU, 1 = ReLU
__global__ void norm_act_kernel(float* __restrict__ x, int n,
                                const float* __restrict__ sums, const float* __restrict__ sumsq,
                                const float* __restrict__ g, const float* __restrict__ be,
                                int actMode) {
  long idx = (long)blockIdx.x * blockDim.x + threadIdx.x;
  if (idx >= (long)n * HID_) return;
  int c = idx & (HID_ - 1);
  float inv_n = 1.f / (float)n;
  float mean = sums[c] * inv_n;
  float var  = sumsq[c] * inv_n - mean * mean;
  float v = (x[idx] - mean) * rsqrtf(var + EPS_) * g[c] + be[c];
  v = (actMode == 0) ? ((v > 0.f) ? v : (__expf(v) - 1.f)) : fmaxf(v, 0.f);
  x[idx] = v;
}

// ============================ host ============================
extern "C" void kernel_launch(void* const* d_in, const int* in_sizes, int n_in,
                              void* d_out, int out_size, void* d_ws, size_t ws_size,
                              hipStream_t stream) {
  const float* x   = (const float*)d_in[0];
  const float* W0  = (const float*)d_in[1];
  const float* al0 = (const float*)d_in[2];
  const float* ar0 = (const float*)d_in[3];
  const float* b0  = (const float*)d_in[4];
  const float* Ws0 = (const float*)d_in[5];
  const float* bs0 = (const float*)d_in[6];
  const float* g0  = (const float*)d_in[7];
  const float* be0 = (const float*)d_in[8];
  const float* W1  = (const float*)d_in[9];
  const float* al1 = (const float*)d_in[10];
  const float* ar1 = (const float*)d_in[11];
  const float* b1  = (const float*)d_in[12];
  const float* Ws1 = (const float*)d_in[13];
  const float* bs1 = (const float*)d_in[14];
  const float* g1  = (const float*)d_in[15];
  const float* be1 = (const float*)d_in[16];
  const float* Wm1 = (const float*)d_in[17];
  const float* bm1 = (const float*)d_in[18];
  const float* gm  = (const float*)d_in[19];
  const float* bem = (const float*)d_in[20];
  const float* Wm2 = (const float*)d_in[21];
  const float* bm2 = (const float*)d_in[22];
  const int* src0  = (const int*)d_in[23];
  const int* dst0  = (const int*)d_in[24];
  const int* src1  = (const int*)d_in[25];
  const int* dst1  = (const int*)d_in[26];
  float* outp = (float*)d_out;

  // --- bump allocator on d_ws ---
  char* base = (char*)d_ws;
  size_t off = 0;
  auto alloc = [&](size_t bytes) -> void* {
    void* p = base + off;
    off += (bytes + 255) & ~(size_t)255;
    return p;
  };
  const int FW0 = 8 * 8 * 512;   // frags per W0 type   (K=256 -> 8 ktiles, N=128 -> 8 ntiles)
  const int FW1 = 8 * 4 * 512;   // frags per W1 type   (K=128)
  const int FM2 = 10 * 4 * 512;  // Wm2: N=153 -> 10 ntiles (padded)
  unsigned short* fW0h  = (unsigned short*)alloc((size_t)T_ * FW0 * 2);
  unsigned short* fW0l  = (unsigned short*)alloc((size_t)T_ * FW0 * 2);
  unsigned short* fWs0h = (unsigned short*)alloc((size_t)FW0 * 2);
  unsigned short* fWs0l = (unsigned short*)alloc((size_t)FW0 * 2);
  unsigned short* fW1h  = (unsigned short*)alloc((size_t)T_ * FW1 * 2);
  unsigned short* fW1l  = (unsigned short*)alloc((size_t)T_ * FW1 * 2);
  unsigned short* fWs1h = (unsigned short*)alloc((size_t)FW1 * 2);
  unsigned short* fWs1l = (unsigned short*)alloc((size_t)FW1 * 2);
  unsigned short* fWm1h = (unsigned short*)alloc((size_t)FW1 * 2);
  unsigned short* fWm1l = (unsigned short*)alloc((size_t)FW1 * 2);
  unsigned short* fWm2h = (unsigned short*)alloc((size_t)FM2 * 2);
  unsigned short* fWm2l = (unsigned short*)alloc((size_t)FM2 * 2);
  float* bias0 = (float*)alloc(HID_ * 4);
  float* bias1 = (float*)alloc(HID_ * 4);
  float* hbuf  = (float*)alloc((size_t)N_SRC0_ * HID_ * 4);   // per-type h = x @ W[j]
  float* out0  = (float*)alloc((size_t)N_DST0_ * HID_ * 4);   // layer0 acc -> layer1 input
  float* out1  = (float*)alloc((size_t)N_DST1_ * HID_ * 4);   // layer1 acc -> MLP input
  float* mh    = (float*)alloc((size_t)N_DST1_ * HID_ * 4);   // MLP hidden
  float* el    = (float*)alloc((size_t)N_SRC0_ * H_ * 4);
  float* er    = (float*)alloc((size_t)N_DST0_ * H_ * 4);
  unsigned* mOrd = (unsigned*)alloc((size_t)N_DST0_ * H_ * 4);
  float* ssum  = (float*)alloc((size_t)N_DST0_ * H_ * 4);
  float* exbuf = (float*)alloc((size_t)E0_ * H_ * 4);
  float* sums  = (float*)alloc(HID_ * 4);
  float* sumsq = (float*)alloc(HID_ * 4);
  (void)ws_size; (void)n_in; (void)in_sizes; (void)out_size;

  auto frags = [&](const float* W, int K, int N, int kt, int nt,
                   unsigned short* h, unsigned short* l) {
    int total = nt * kt * 512;
    make_frags<<<(total + 255) / 256, 256, 0, stream>>>(W, K, N, kt, nt, h, l);
  };
  // --- weight prep ---
  for (int j = 0; j < T_; ++j)
    frags(W0 + (size_t)j * IN_ * HID_, IN_, HID_, 8, 8, fW0h + (size_t)j * FW0, fW0l + (size_t)j * FW0);
  frags(Ws0, IN_, HID_, 8, 8, fWs0h, fWs0l);
  for (int j = 0; j < T_; ++j)
    frags(W1 + (size_t)j * HID_ * HID_, HID_, HID_, 4, 8, fW1h + (size_t)j * FW1, fW1l + (size_t)j * FW1);
  frags(Ws1, HID_, HID_, 4, 8, fWs1h, fWs1l);
  frags(Wm1, HID_, HID_, 4, 8, fWm1h, fWm1l);
  frags(Wm2, HID_, OUT_, 4, 10, fWm2h, fWm2l);
  bias_total_kernel<<<1, HID_, 0, stream>>>(bs0, b0, bias0);
  bias_total_kernel<<<1, HID_, 0, stream>>>(bs1, b1, bias1);

  auto zero = [&](void* p, long words) {
    fill_u32<<<256, 256, 0, stream>>>((unsigned*)p, words, 0u);
  };
  auto gemm = [&](const float* A, int M, int K,
                  const unsigned short* fh, const unsigned short* fl,
                  int ntiles, int N, const float* bias, float* C) {
    gemm_wmma_bf16x3<<<dim3(M / 16, (ntiles + 7) / 8), 256, 0, stream>>>(
        A, K, fh, fl, ntiles, N, bias, C);
  };

  auto run_layer = [&](const float* xin, int n_src, int n_dst, int K, int E,
                       const unsigned short* fWh, const unsigned short* fWl, int perW,
                       const unsigned short* fWsh, const unsigned short* fWsl,
                       const float* biasTot, const float* al, const float* ar,
                       const int* src, const int* dst,
                       const float* g, const float* be, float* outBuf) {
    // out = x_dst @ Ws + (bs + sum_j b[j])
    gemm(xin, n_dst, K, fWsh, fWsl, 8, HID_, biasTot, outBuf);
    for (int j = 0; j < T_; ++j) {
      gemm(xin, n_src, K, fWh + (size_t)j * perW, fWl + (size_t)j * perW, 8, HID_, nullptr, hbuf);
      el_er_kernel<<<(n_src * H_ + 255) / 256, 256, 0, stream>>>(
          hbuf, n_src, n_dst, al + j * H_ * D_, ar + j * H_ * D_, el, er);
      zero(mOrd, (long)n_dst * H_);
      zero(ssum, (long)n_dst * H_);
      edge_max_kernel<<<(E + 255) / 256, 256, 0, stream>>>(
          src + (size_t)j * E, dst + (size_t)j * E, E, el, er, mOrd);
      edge_sum_kernel<<<(E + 255) / 256, 256, 0, stream>>>(
          src + (size_t)j * E, dst + (size_t)j * E, E, el, er, mOrd, exbuf, ssum);
      edge_msg_kernel<<<(int)(((long)E * 32 + 255) / 256), 256, 0, stream>>>(
          src + (size_t)j * E, dst + (size_t)j * E, E, exbuf, ssum, hbuf, outBuf);
    }
    zero(sums, HID_); zero(sumsq, HID_);
    colstats_kernel<<<256, HID_, 0, stream>>>(outBuf, n_dst, sums, sumsq);
    norm_act_kernel<<<(int)(((long)n_dst * HID_ + 255) / 256), 256, 0, stream>>>(
        outBuf, n_dst, sums, sumsq, g, be, /*ELU*/0);
  };

  // ---- layer 0 ----
  run_layer(x, N_SRC0_, N_DST0_, IN_, E0_, fW0h, fW0l, FW0, fWs0h, fWs0l,
            bias0, al0, ar0, src0, dst0, g0, be0, out0);
  // ---- layer 1 ----
  run_layer(out0, N_DST0_, N_DST1_, HID_, E1_, fW1h, fW1l, FW1, fWs1h, fWs1l,
            bias1, al1, ar1, src1, dst1, g1, be1, out1);

  // ---- MLP ----
  gemm(out1, N_DST1_, HID_, fWm1h, fWm1l, 8, HID_, bm1, mh);
  zero(sums, HID_); zero(sumsq, HID_);
  colstats_kernel<<<256, HID_, 0, stream>>>(mh, N_DST1_, sums, sumsq);
  norm_act_kernel<<<(int)(((long)N_DST1_ * HID_ + 255) / 256), 256, 0, stream>>>(
      mh, N_DST1_, sums, sumsq, gm, bem, /*ReLU*/1);
  gemm(mh, N_DST1_, HID_, fWm2h, fWm2l, 10, OUT_, bm2, outp);
}
